// MultiHeadSelfAttention_4827543241224
// MI455X (gfx1250) — compile-verified, matrix operand order checked by reading, not run on hardware
//
#include <hip/hip_runtime.h>
#include <hip/hip_bf16.h>

// Problem constants (reference: B=4, T=2048, C=1024, H=16, D=64)
#define BB 4
#define TT 2048
#define CC 1024
#define HH 16
#define DD 64

typedef __attribute__((ext_vector_type(8)))  float          v8f;
typedef __attribute__((ext_vector_type(16))) __bf16         v16bf;
typedef __attribute__((ext_vector_type(8)))  unsigned short ushort8;

struct Frag16 { ushort8 lo, hi; };
union  FragU  { Frag16 f; v16bf v; };

__device__ __forceinline__ unsigned short f2bf(float x) {
  unsigned int u = __builtin_bit_cast(unsigned int, x);
  u += 0x7FFFu + ((u >> 16) & 1u);           // round-to-nearest-even
  return (unsigned short)(u >> 16);
}

__device__ __forceinline__ v16bf make_frag(ushort8 lo, ushort8 hi) {
  FragU u; u.f.lo = lo; u.f.hi = hi; return u.v;
}

__device__ __forceinline__ v8f wmma_bf16(v16bf a, v16bf b, v8f c) {
  // 8 args: (neg_a, A, neg_b, B, c_mod, C, reuse_a, reuse_b)
  return __builtin_amdgcn_wmma_f32_16x16x32_bf16(false, a, false, b, (short)0, c,
                                                 false, false);
}

// CDNA5 async global->LDS copy (per-lane b128, tracked with ASYNCcnt).
// vdst = 32-bit LDS byte offset (low 32 bits of the flat shared address),
// vaddr = 64-bit global address. Inline asm keeps it portable across the
// clang-22 / clang-23 builtin-arity split.
__device__ __forceinline__ void async_copy_b128(const void* gptr, void* lptr) {
  unsigned long long ga = (unsigned long long)(size_t)gptr;
  unsigned int       la = (unsigned int)(size_t)lptr;  // LDS aperture low bits
  asm volatile("global_load_async_to_lds_b128 %0, %1, off"
               :: "v"(la), "v"(ga)
               : "memory");
}
__device__ __forceinline__ void wait_async0() {
  asm volatile("s_wait_asynccnt 0x0" ::: "memory");
}

// ---------------------------------------------------------------------------
// f32 -> bf16 elementwise conversion (grid-stride)
// ---------------------------------------------------------------------------
__global__ void f32_to_bf16_kernel(const float* __restrict__ in,
                                   unsigned short* __restrict__ out, int n) {
  int i = blockIdx.x * blockDim.x + threadIdx.x;
  int stride = gridDim.x * blockDim.x;
  for (; i < n; i += stride) out[i] = f2bf(in[i]);
}

// ---------------------------------------------------------------------------
// bf16 GEMM: out(MxN) = A(MxK, row-major bf16) * B(KxN, row-major bf16)
// Block tile 64x64, 4 waves, each wave owns 2x2 of 16x16 WMMA tiles, K-step 32.
// A-tile staged with async global->LDS DMA; B-tile transposed through VGPRs.
// F32OUT: write f32 + bias; else write bf16.
// ---------------------------------------------------------------------------
template <bool F32OUT>
__global__ __launch_bounds__(128)
void gemm_bf16_kernel(const unsigned short* __restrict__ A,
                      const unsigned short* __restrict__ Bm,
                      const float* __restrict__ bias,
                      float* __restrict__ outF,
                      unsigned short* __restrict__ outB,
                      int M, int N, int K) {
  __shared__ __align__(16) unsigned short lA[64][40];  // [m][k], pad to 40
  __shared__ __align__(16) unsigned short lB[64][40];  // [n][k] (transposed)

  const int tid  = threadIdx.x;
  const int lane = tid & 31;
  const int wid  = tid >> 5;
  const int g    = lane >> 4;      // half-wave group
  const int ln16 = lane & 15;
  const int m0   = blockIdx.y * 64;
  const int n0   = blockIdx.x * 64;
  const int wm   = (wid >> 1) * 32;
  const int wn   = (wid & 1) * 32;

  v8f zero = {0.f, 0.f, 0.f, 0.f, 0.f, 0.f, 0.f, 0.f};
  v8f acc[2][2];
#pragma unroll
  for (int i = 0; i < 2; i++)
#pragma unroll
    for (int j = 0; j < 2; j++) acc[i][j] = zero;

  for (int k0 = 0; k0 < K; k0 += 32) {
    __syncthreads();
    // Stage A tile (64x32) via async DMA; B tile (32x64) transposed to [n][k].
#pragma unroll
    for (int it = 0; it < 2; it++) {
      int task = tid + it * 128;
      // A: 64 rows x 4 segments of 8 bf16 -> async b128 straight into LDS
      int r = task >> 2, seg = task & 3;
      async_copy_b128(A + (size_t)(m0 + r) * K + k0 + seg * 8,
                      &lA[r][seg * 8]);
      // B: 32 k-rows x 8 segments of 8 bf16, scatter transposed
      int kk = task >> 3, s2 = task & 7;
      ushort8 vb = *(const ushort8*)(Bm + (size_t)(k0 + kk) * N + n0 + s2 * 8);
#pragma unroll
      for (int e = 0; e < 8; e++) lB[s2 * 8 + e][kk] = vb[e];
    }
    wait_async0();          // async LDS writes visible before the barrier
    __syncthreads();

    v16bf af[2], bf[2];
#pragma unroll
    for (int t = 0; t < 2; t++) {
      int row = wm + t * 16 + ln16;
      af[t] = make_frag(*(const ushort8*)&lA[row][g * 8],
                        *(const ushort8*)&lA[row][16 + g * 8]);
      int nrow = wn + t * 16 + ln16;
      bf[t] = make_frag(*(const ushort8*)&lB[nrow][g * 8],
                        *(const ushort8*)&lB[nrow][16 + g * 8]);
    }
#pragma unroll
    for (int i = 0; i < 2; i++)
#pragma unroll
      for (int j = 0; j < 2; j++) acc[i][j] = wmma_bf16(af[i], bf[j], acc[i][j]);
  }

  // Epilogue: C/D layout: VGPR r -> row (r + 8*g), col = lane%16.
#pragma unroll
  for (int i = 0; i < 2; i++)
#pragma unroll
    for (int j = 0; j < 2; j++)
#pragma unroll
      for (int r = 0; r < 8; r++) {
        int m = m0 + wm + i * 16 + r + 8 * g;
        int n = n0 + wn + j * 16 + ln16;
        float v = acc[i][j][r];
        if constexpr (F32OUT) {
          outF[(size_t)m * N + n] = v + bias[n];
        } else {
          outB[(size_t)m * N + n] = f2bf(v);
        }
      }
}

// ---------------------------------------------------------------------------
// Flash attention over bf16 QKV buffer laid out (B*T, 3C) row-major:
//   Q at col h*64, K at C + h*64, V at 2C + h*64.
// Workgroup: 4 waves, each wave = one 16-row query tile (64 queries/WG),
// shared K/V staging per 32-key tile (K via async DMA). Online softmax with
// the padding mask folded into an additive fma bias (one cndmask per tile).
// Writes bf16 attention output (B*T, C).
// ---------------------------------------------------------------------------
__global__ __launch_bounds__(128)
void flash_attn_kernel(const unsigned short* __restrict__ qkv,
                       const unsigned char* __restrict__ pmask,
                       unsigned short* __restrict__ aout) {
  __shared__ __align__(16) unsigned short lK[32][72];     // [key][d], pad 72
  __shared__ __align__(16) unsigned short lV[64][40];     // [d][key] (transposed)
  __shared__ __align__(16) unsigned short lP[4][16][32];  // per-wave P bounce

  const int tid  = threadIdx.x;
  const int lane = tid & 31;
  const int wid  = tid >> 5;
  const int g    = lane >> 4;
  const int ln16 = lane & 15;

  const int b  = blockIdx.y / HH;
  const int h  = blockIdx.y % HH;
  const int q0 = blockIdx.x * 64 + wid * 16;

  const size_t rs = 3 * CC;  // qkv row stride (elements)
  const unsigned short* Qp = qkv + (size_t)(b * TT) * rs + h * DD;
  const unsigned short* Kp = Qp + CC;
  const unsigned short* Vp = Qp + 2 * CC;

  // Q fragments: two 16x32 A-operands covering D=64 contraction.
  v16bf qf[2];
  {
    const unsigned short* qrow = Qp + (size_t)(q0 + ln16) * rs;
#pragma unroll
    for (int c = 0; c < 2; c++)
      qf[c] = make_frag(*(const ushort8*)(qrow + c * 32 + g * 8),
                        *(const ushort8*)(qrow + c * 32 + 16 + g * 8));
  }

  v8f zero = {0.f, 0.f, 0.f, 0.f, 0.f, 0.f, 0.f, 0.f};
  v8f o[4];
#pragma unroll
  for (int c = 0; c < 4; c++) o[c] = zero;
  float mrow[8], lrow[8];
#pragma unroll
  for (int j = 0; j < 8; j++) { mrow[j] = -1e30f; lrow[j] = 0.f; }

  const float scale = 0.125f;  // D^-0.5

  for (int kt = 0; kt < TT; kt += 32) {
    __syncthreads();
    // Stage K ([key][d], async DMA) and V (transposed [d][key] via VGPRs).
#pragma unroll
    for (int it = 0; it < 2; it++) {
      int task = tid + it * 128;
      int kk = task >> 3, seg = task & 7;
      async_copy_b128(Kp + (size_t)(kt + kk) * rs + seg * 8,
                      &lK[kk][seg * 8]);
      const unsigned short* vrow = Vp + (size_t)(kt + kk) * rs;
      ushort8 vv = *(const ushort8*)(vrow + seg * 8);
#pragma unroll
      for (int e = 0; e < 8; e++) lV[seg * 8 + e][kk] = vv[e];
    }
    // Prefetch next key tile while this one is consumed.
    if (kt + 32 < TT) {
      const unsigned short* nk = Kp + (size_t)(kt + 32 + (tid >> 2)) * rs;
      __builtin_prefetch(nk, 0, 1);
      const unsigned short* nv = Vp + (size_t)(kt + 32 + (tid >> 2)) * rs;
      __builtin_prefetch(nv, 0, 1);
    }
    wait_async0();          // async LDS writes visible before the barrier
    __syncthreads();

    // Padding mask for this lane's two key columns, as an additive fma bias
    // (mask is per-column => one select per tile, not per element).
    float bias0 = pmask[(size_t)b * TT + kt + ln16]      ? 0.f : -1e30f;
    float bias1 = pmask[(size_t)b * TT + kt + 16 + ln16] ? 0.f : -1e30f;

    // Scores: S0 = keys [0,16), S1 = keys [16,32); 2 d-chunks each.
    v8f s0 = zero, s1 = zero;
#pragma unroll
    for (int c = 0; c < 2; c++) {
      v16bf kA = make_frag(*(const ushort8*)&lK[ln16][c * 32 + g * 8],
                           *(const ushort8*)&lK[ln16][c * 32 + 16 + g * 8]);
      s0 = wmma_bf16(qf[c], kA, s0);
      v16bf kB = make_frag(*(const ushort8*)&lK[16 + ln16][c * 32 + g * 8],
                           *(const ushort8*)&lK[16 + ln16][c * 32 + 16 + g * 8]);
      s1 = wmma_bf16(qf[c], kB, s1);
    }

    // Online softmax. Each (VGPR j, half-wave g) holds row m = j + 8*g;
    // the 16 lanes of a half-wave hold the 16 columns -> xor reductions <=8
    // stay inside the group.
#pragma unroll
    for (int j = 0; j < 8; j++) {
      float a0 = fmaf(s0[j], scale, bias0);
      float a1 = fmaf(s1[j], scale, bias1);
      float tm = fmaxf(a0, a1);
#pragma unroll
      for (int off = 1; off < 16; off <<= 1)
        tm = fmaxf(tm, __shfl_xor(tm, off, 32));
      float mn    = fmaxf(mrow[j], tm);
      float alpha = __expf(mrow[j] - mn);
      float p0    = __expf(a0 - mn);
      float p1    = __expf(a1 - mn);
      float rsum  = p0 + p1;
#pragma unroll
      for (int off = 1; off < 16; off <<= 1) rsum += __shfl_xor(rsum, off, 32);
      lrow[j] = lrow[j] * alpha + rsum;
      mrow[j] = mn;
#pragma unroll
      for (int c = 0; c < 4; c++) o[c][j] *= alpha;
      // C-layout -> LDS (row-major [m][key]) for A-layout reload.
      lP[wid][j + 8 * g][ln16]      = f2bf(p0);
      lP[wid][j + 8 * g][16 + ln16] = f2bf(p1);
    }

    // Ensure per-wave LDS stores land before the A-fragment reload.
    asm volatile("s_wait_dscnt 0x0" ::: "memory");

    v16bf pf = make_frag(*(const ushort8*)&lP[wid][ln16][g * 8],
                         *(const ushort8*)&lP[wid][ln16][16 + g * 8]);
#pragma unroll
    for (int c = 0; c < 4; c++) {
      int drow = c * 16 + ln16;
      v16bf vf = make_frag(*(const ushort8*)&lV[drow][g * 8],
                           *(const ushort8*)&lV[drow][16 + g * 8]);
      o[c] = wmma_bf16(pf, vf, o[c]);
    }
  }

  // Normalize and store bf16 attention output (B*T, C).
#pragma unroll
  for (int j = 0; j < 8; j++) {
    float inv = 1.0f / lrow[j];
    int t = q0 + j + 8 * g;
    size_t base = (size_t)(b * TT + t) * CC + h * DD;
#pragma unroll
    for (int c = 0; c < 4; c++)
      aout[base + c * 16 + ln16] = f2bf(o[c][j] * inv);
  }
}

// ---------------------------------------------------------------------------
// Launcher. Inputs: x(f32 B,T,C), padding_mask(bool B,T), w_qkv(f32 C,3C),
// w_out(f32 C,C), b_out(f32 C). Output: f32 (B,T,C).
// ---------------------------------------------------------------------------
extern "C" void kernel_launch(void* const* d_in, const int* in_sizes, int n_in,
                              void* d_out, int out_size, void* d_ws, size_t ws_size,
                              hipStream_t stream) {
  const float*         x     = (const float*)d_in[0];
  const unsigned char* pmask = (const unsigned char*)d_in[1];  // jnp bool_ = 1B
  const float*         w_qkv = (const float*)d_in[2];
  const float*         w_out = (const float*)d_in[3];
  const float*         b_out = (const float*)d_in[4];
  float*               out   = (float*)d_out;

  char* ws = (char*)d_ws;
  size_t off = 0;
  auto carve = [&](size_t bytes) {
    size_t p = off;
    off += (bytes + 255) & ~(size_t)255;
    return p;
  };
  unsigned short* xbf    = (unsigned short*)(ws + carve((size_t)BB * TT * CC * 2));
  unsigned short* wqkvbf = (unsigned short*)(ws + carve((size_t)CC * 3 * CC * 2));
  unsigned short* woutbf = (unsigned short*)(ws + carve((size_t)CC * CC * 2));
  unsigned short* qkvbf  = (unsigned short*)(ws + carve((size_t)BB * TT * 3 * CC * 2));
  unsigned short* attbf  = (unsigned short*)(ws + carve((size_t)BB * TT * CC * 2));

  // 1) Down-convert inputs to bf16 once.
  f32_to_bf16_kernel<<<dim3(4096), dim3(256), 0, stream>>>(x, xbf, BB * TT * CC);
  f32_to_bf16_kernel<<<dim3(2048), dim3(256), 0, stream>>>(w_qkv, wqkvbf, CC * 3 * CC);
  f32_to_bf16_kernel<<<dim3(1024), dim3(256), 0, stream>>>(w_out, woutbf, CC * CC);

  // 2) QKV projection: (8192x1024) @ (1024x3072) -> bf16.
  gemm_bf16_kernel<false><<<dim3(3 * CC / 64, BB * TT / 64), dim3(128), 0, stream>>>(
      xbf, wqkvbf, nullptr, nullptr, qkvbf, BB * TT, 3 * CC, CC);

  // 3) Flash attention per (b,h), 64 queries per workgroup.
  flash_attn_kernel<<<dim3(TT / 64, BB * HH), dim3(128), 0, stream>>>(
      qkvbf, pmask, attbf);

  // 4) Output projection + bias: (8192x1024) @ (1024x1024) -> f32 out.
  gemm_bf16_kernel<true><<<dim3(CC / 64, BB * TT / 64), dim3(128), 0, stream>>>(
      attbf, woutbf, b_out, out, nullptr, BB * TT, CC, CC);

  (void)in_sizes; (void)n_in; (void)out_size; (void)ws_size;
}
